// BigbirdSimulatedAttention_32057635897446
// MI455X (gfx1250) — matrix operand order repaired and checked by reading, not compiled, not run on hardware
//
#include <hip/hip_runtime.h>
#include <hip/hip_bf16.h>

#define SEQ   4096
#define HEADS 12
#define DHEAD 64
#define BLK   64
#define NB    (SEQ / BLK)
#define KP    72   // LDS row pitch in halves (64 + 8 pad = 144B rows; TDM pad reproduces this)

typedef __attribute__((ext_vector_type(16))) _Float16 v16h;
typedef __attribute__((ext_vector_type(8)))  _Float16 v8h;
typedef __attribute__((ext_vector_type(8)))  float    v8f;
typedef __attribute__((ext_vector_type(4)))  unsigned int u32x4;
typedef __attribute__((ext_vector_type(8)))  int      i32x8;
typedef __attribute__((ext_vector_type(4)))  int      i32x4;

#if defined(__has_builtin)
#if __has_builtin(__builtin_amdgcn_tensor_load_to_lds) && __has_builtin(__builtin_amdgcn_s_wait_tensorcnt)
#define USE_TDM 1
#endif
#endif
#ifndef USE_TDM
#define USE_TDM 0
#endif

static __device__ __forceinline__ v16h pack16(v8h lo, v8h hi) {
  v16h r;
#pragma unroll
  for (int i = 0; i < 8; ++i) { r[i] = lo[i]; r[8 + i] = hi[i]; }
  return r;
}

// B-matrix fragment (32x16 f16): each lane reads 16 contiguous halves (2 x ds_load_b128).
static __device__ __forceinline__ v16h ld_bfrag(const _Float16* p) {
  return pack16(*(const v8h*)p, *(const v8h*)(p + 8));
}

// A-matrix fragment (16x32 f16) from a row: halves at dk+16*hl .. dk+16*hl+15.
static __device__ __forceinline__ v16h ld_afrag(const _Float16* row, int dk, int hl) {
  return pack16(*(const v8h*)(row + dk + 16 * hl),
                *(const v8h*)(row + dk + 16 * hl + 8));
}

static __device__ __forceinline__ v8f wmma_f16(v16h a, v16h b, v8f c) {
  return __builtin_amdgcn_wmma_f32_16x16x32_f16(false, a, false, b, (short)0, c,
                                                false, false);
}

static __device__ __forceinline__ unsigned int lds_off(const void* p) {
  // Generic LDS address low 32 bits == LDS byte offset (aperture scheme, ISA 10.2).
  return (unsigned int)(uintptr_t)p;
}

#if USE_TDM
// Issue one TDM 64x64 f16 tile copy: global (row pitch 64 halves, contiguous tile)
// -> LDS with a 16B pad after every 128B row => LDS pitch 72 halves (KP).
static __device__ __forceinline__ void tdm_tile_64x64_f16(unsigned int lds_byte_off,
                                                          const _Float16* gsrc) {
  unsigned long long ga = (unsigned long long)(uintptr_t)gsrc;
  u32x4 g0;
  g0[0] = 1u;                                    // count=1 (valid user descriptor)
  g0[1] = lds_byte_off;                          // lds_addr
  g0[2] = (unsigned int)ga;                      // global_addr[31:0]
  g0[3] = (unsigned int)((ga >> 32) & 0x01FFFFFFu) | 0x80000000u; // addr[56:32] | type=2
  i32x8 g1;
  g1[0] = (1 << 16)      // data_size = 2 bytes
        | (1 << 20)      // pad_enable
        | (4 << 22)      // pad_interval code 4: every 32 DWORDs (128B row)
        | (3 << 25);     // pad_amount  code 3: 4 DWORDs (16B) -> 72-half pitch
  g1[1] = (64 << 16);    // tensor_dim0 = 64 (tile == tensor: no OOB)
  g1[2] = (64 << 16);    // tensor_dim1 = 64
  g1[3] = (64 << 16);    // tile_dim0 = 64
  g1[4] = 64;            // tile_dim1 = 64, tile_dim2 = 0
  g1[5] = 64;            // tensor_dim0_stride = 64
  g1[6] = 0;
  g1[7] = 0;
  i32x4 z4 = {};
#if defined(__clang_major__) && (__clang_major__ >= 23)
  i32x8 z8 = {};
  __builtin_amdgcn_tensor_load_to_lds(g0, g1, z4, z4, z8, 0);
#else
  __builtin_amdgcn_tensor_load_to_lds(g0, g1, z4, z4, 0);
#endif
}
#endif

// ---------------------------------------------------------------------------
// Prep pass: fp32 -> f16 once (amortized; ~57MB traffic @ 23.3TB/s ~ 2.5us).
// qf = q * (1/sqrt(D)) row-major; kf = k row-major; vt = per-64-block transpose
// [h][block][d][key] so V B-fragments become contiguous 16-half reads.
// ---------------------------------------------------------------------------
__global__ __launch_bounds__(256)
void bigbird_prep_f16_kernel(const float* __restrict__ q, const float* __restrict__ k,
                             const float* __restrict__ v, _Float16* __restrict__ qf,
                             _Float16* __restrict__ kf, _Float16* __restrict__ vt) {
  int i = blockIdx.x * 256 + threadIdx.x;
  if (i >= HEADS * SEQ * DHEAD) return;
  qf[i] = (_Float16)(q[i] * 0.125f);
  kf[i] = (_Float16)k[i];
  int d = i & 63;
  int s = (i >> 6) & (SEQ - 1);
  int h = i >> 18;                       // SEQ*DHEAD = 2^18
  int b = s >> 6, key = s & 63;
  vt[(((size_t)h * NB + b) * DHEAD + d) * BLK + key] = (_Float16)v[i];
}

// ---------------------------------------------------------------------------
// Main kernel: block-sparse flash attention, f16 tiles, TDM double buffering.
// ---------------------------------------------------------------------------
__global__ __launch_bounds__(128)
void bigbird_attn_tdm_kernel(const _Float16* __restrict__ qf,
                             const _Float16* __restrict__ kf,
                             const _Float16* __restrict__ vt,
                             const int* __restrict__ mask,
                             float* __restrict__ out) {
  __shared__ alignas(32) _Float16 kbuf[2][BLK * KP];    // K tile f16, [key][d]
  __shared__ alignas(32) _Float16 vbuf[2][DHEAD * KP];  // V^T tile f16, [d][key]
  __shared__ alignas(32) _Float16 psm[4 * 16 * KP];     // per-wave probs strip
  __shared__ int flags[NB];
  __shared__ int act[NB];
  __shared__ int nact_s;

  const int h    = blockIdx.y;
  const int qb   = blockIdx.x;
  const int tid  = threadIdx.x;
  const int lane = tid & 31;
  const int w    = tid >> 5;
  const int hl   = lane >> 4;
  const int ln   = lane & 15;

  // ---- block-constant BigBird mask: one int per 64x64 tile (16KB total vs 67MB dense).
  if (tid < NB) flags[tid] = mask[(size_t)(qb * BLK) * SEQ + (size_t)tid * BLK];
  __syncthreads();
  if (tid == 0) {
    int n = 0;
#pragma unroll 4
    for (int b = 0; b < NB; ++b)
      if (flags[b]) act[n++] = b;
    nact_s = n;
  }
  __syncthreads();
  const int nact = nact_s;

  // ---- Q: this wave's 16 rows as two 16x32 f16 A fragments (pre-scaled in prep).
  const _Float16* __restrict__ qrow =
      qf + ((size_t)h * SEQ + (size_t)(qb * BLK + w * 16 + ln)) * DHEAD;
  v16h qa0 = pack16(*(const v8h*)(qrow + 16 * hl), *(const v8h*)(qrow + 8 + 16 * hl));
  v16h qa1 = pack16(*(const v8h*)(qrow + 32 + 16 * hl), *(const v8h*)(qrow + 40 + 16 * hl));

  float mrun[8], lrun[8];
  v8f o[4];
  {
    v8f z = {};
#pragma unroll
    for (int t = 0; t < 4; ++t) o[t] = z;
#pragma unroll
    for (int r = 0; r < 8; ++r) { mrun[r] = -INFINITY; lrun[r] = 0.f; }
  }

  const _Float16* __restrict__ Kh = kf + ((size_t)h * SEQ) * DHEAD;
  const _Float16* __restrict__ Vh = vt + ((size_t)h * NB) * DHEAD * BLK;
  _Float16* __restrict__ pw = psm + w * 16 * KP;

#if USE_TDM
  // Prologue: wave 0 DMAs the first tile pair (EXEC-independent, 1 issue/wave).
  if (w == 0 && nact > 0) {
    tdm_tile_64x64_f16(lds_off(kbuf[0]), Kh + (size_t)act[0] * BLK * DHEAD);
    tdm_tile_64x64_f16(lds_off(vbuf[0]), Vh + (size_t)act[0] * DHEAD * BLK);
  }
#endif

  int buf = 0;
  for (int j = 0; j < nact; ++j) {
#if USE_TDM
    if (w == 0) {
      if (j + 1 < nact) {  // software pipeline: prefetch next tile pair into other buffer
        tdm_tile_64x64_f16(lds_off(kbuf[buf ^ 1]), Kh + (size_t)act[j + 1] * BLK * DHEAD);
        tdm_tile_64x64_f16(lds_off(vbuf[buf ^ 1]), Vh + (size_t)act[j + 1] * DHEAD * BLK);
        __builtin_amdgcn_s_wait_tensorcnt(2);  // current pair done; next pair in flight
      } else {
        __builtin_amdgcn_s_wait_tensorcnt(0);
      }
    }
    __syncthreads();  // TDM-written LDS now visible to all waves
#else
    // Fallback: cooperative f16 copy (still halves bytes vs fp32 staging).
    {
      const int kb = act[j];
      const _Float16* gK = Kh + (size_t)kb * BLK * DHEAD;
      const _Float16* gV = Vh + (size_t)kb * DHEAD * BLK;
      __syncthreads();
#pragma unroll 4
      for (int i = tid; i < BLK * DHEAD; i += 128) {
        int row = i >> 6, c = i & 63;
        kbuf[buf][row * KP + c] = gK[i];
        vbuf[buf][row * KP + c] = gV[i];
      }
      __syncthreads();
    }
#endif

    const _Float16* __restrict__ ksm = kbuf[buf];
    const _Float16* __restrict__ vsm = vbuf[buf];

    // ---- S = Q K^T : four 16x16 tiles per wave, K-dim 64 = 2 chained WMMAs each.
    v8f s[4];
#pragma unroll
    for (int t = 0; t < 4; ++t) {
      const _Float16* kr = &ksm[(16 * t + ln) * KP];
      v8f c = {};
      c = wmma_f16(qa0, ld_bfrag(kr + 16 * hl), c);
      c = wmma_f16(qa1, ld_bfrag(kr + 32 + 16 * hl), c);
      s[t] = c;
    }

    // ---- online softmax over this 16x64 strip (rows live across 16-lane halves).
    float mblk[8];
#pragma unroll
    for (int r = 0; r < 8; ++r) {
      float mv = fmaxf(fmaxf(s[0][r], s[1][r]), fmaxf(s[2][r], s[3][r]));
      mblk[r] = mv;
    }
#pragma unroll
    for (int d = 1; d < 16; d <<= 1)
#pragma unroll
      for (int r = 0; r < 8; ++r)
        mblk[r] = fmaxf(mblk[r], __shfl_xor(mblk[r], d, 16));

    float alpha[8], lblk[8];
#pragma unroll
    for (int r = 0; r < 8; ++r) {
      float mnew = fmaxf(mrun[r], mblk[r]);
      alpha[r] = __expf(mrun[r] - mnew);
      mrun[r]  = mnew;
      lblk[r]  = 0.f;
    }
#pragma unroll
    for (int t = 0; t < 4; ++t)
#pragma unroll
      for (int r = 0; r < 8; ++r) {
        float p = __expf(s[t][r] - mrun[r]);
        s[t][r] = p;
        lblk[r] += p;
      }
#pragma unroll
    for (int d = 1; d < 16; d <<= 1)
#pragma unroll
      for (int r = 0; r < 8; ++r)
        lblk[r] += __shfl_xor(lblk[r], d, 16);
#pragma unroll
    for (int r = 0; r < 8; ++r) lrun[r] = lrun[r] * alpha[r] + lblk[r];

    // ---- P (f32 C-layout) -> f16 A-layout via wave-private LDS strip.
#pragma unroll
    for (int t = 0; t < 4; ++t)
#pragma unroll
      for (int r = 0; r < 8; ++r)
        pw[(8 * hl + r) * KP + 16 * t + ln] = (_Float16)s[t][r];
    asm volatile("s_wait_dscnt 0" ::: "memory");
    v16h pa0 = ld_afrag(pw + ln * KP, 0, hl);
    v16h pa1 = ld_afrag(pw + ln * KP, 32, hl);

    // ---- O = O*alpha + P V : accumulate straight into the WMMA C operand.
#pragma unroll
    for (int t = 0; t < 4; ++t) {
      const _Float16* vr = &vsm[(16 * t + ln) * KP];
      v8f c = o[t];
#pragma unroll
      for (int r = 0; r < 8; ++r) c[r] *= alpha[r];
      c = wmma_f16(pa0, ld_bfrag(vr + 16 * hl), c);
      c = wmma_f16(pa1, ld_bfrag(vr + 32 + 16 * hl), c);
      o[t] = c;
    }

    __syncthreads();  // all waves done with buf before wave0 overwrites it next iter
    buf ^= 1;
  }

  // ---- epilogue: normalize and store fp32 output.
#pragma unroll
  for (int r = 0; r < 8; ++r) {
    const size_t row = (size_t)h * SEQ + (size_t)(qb * BLK + w * 16 + 8 * hl + r);
    float inv = 1.0f / lrun[r];
#pragma unroll
    for (int t = 0; t < 4; ++t)
      out[row * DHEAD + 16 * t + ln] = o[t][r] * inv;
  }
}

// ---------------------------------------------------------------------------
// Fallback kernel (no workspace needed): fp32 staging with in-kernel conversion.
// ---------------------------------------------------------------------------
__global__ __launch_bounds__(128)
void bigbird_attn_fp32_kernel(const float* __restrict__ q, const float* __restrict__ k,
                              const float* __restrict__ v, const int* __restrict__ mask,
                              float* __restrict__ out) {
  __shared__ alignas(32) _Float16 ksm[BLK * KP];
  __shared__ alignas(32) _Float16 vtsm[DHEAD * KP];
  __shared__ alignas(32) _Float16 psm[4 * 16 * KP];

  const int h = blockIdx.y, qb = blockIdx.x, tid = threadIdx.x;
  const int lane = tid & 31, w = tid >> 5, hl = lane >> 4, ln = lane & 15;

  const float* __restrict__ Kh = k + ((size_t)h * SEQ) * DHEAD;
  const float* __restrict__ Vh = v + ((size_t)h * SEQ) * DHEAD;

  const float scale = 0.125f;
  const float* __restrict__ qrow =
      q + ((size_t)h * SEQ + (size_t)(qb * BLK + w * 16 + ln)) * DHEAD;
  v16h qa0, qa1;
#pragma unroll
  for (int i = 0; i < 8; ++i) {
    qa0[i]     = (_Float16)(qrow[      16 * hl + i] * scale);
    qa0[8 + i] = (_Float16)(qrow[8 +  16 * hl + i] * scale);
    qa1[i]     = (_Float16)(qrow[32 +  16 * hl + i] * scale);
    qa1[8 + i] = (_Float16)(qrow[40 +  16 * hl + i] * scale);
  }

  float mrun[8], lrun[8];
  v8f o[4];
  {
    v8f z = {};
#pragma unroll
    for (int t = 0; t < 4; ++t) o[t] = z;
#pragma unroll
    for (int r = 0; r < 8; ++r) { mrun[r] = -INFINITY; lrun[r] = 0.f; }
  }
  _Float16* __restrict__ pw = psm + w * 16 * KP;

  for (int kb = 0; kb < NB; ++kb) {
    if (mask[(size_t)(qb * BLK) * SEQ + kb * BLK] == 0) continue;
    __syncthreads();
    {
      const float* gK = Kh + (size_t)kb * BLK * DHEAD;
      const float* gV = Vh + (size_t)kb * BLK * DHEAD;
#pragma unroll 4
      for (int i = tid; i < BLK * DHEAD; i += 128) {
        int key = i >> 6, d = i & 63;
        ksm[key * KP + d]  = (_Float16)gK[i];
        vtsm[d * KP + key] = (_Float16)gV[i];
      }
    }
    __syncthreads();

    v8f s[4];
#pragma unroll
    for (int t = 0; t < 4; ++t) {
      const _Float16* kr = &ksm[(16 * t + ln) * KP];
      v8f c = {};
      c = wmma_f16(qa0, ld_bfrag(kr + 16 * hl), c);
      c = wmma_f16(qa1, ld_bfrag(kr + 32 + 16 * hl), c);
      s[t] = c;
    }

    float mblk[8];
#pragma unroll
    for (int r = 0; r < 8; ++r)
      mblk[r] = fmaxf(fmaxf(s[0][r], s[1][r]), fmaxf(s[2][r], s[3][r]));
#pragma unroll
    for (int d = 1; d < 16; d <<= 1)
#pragma unroll
      for (int r = 0; r < 8; ++r)
        mblk[r] = fmaxf(mblk[r], __shfl_xor(mblk[r], d, 16));

    float alpha[8], lblk[8];
#pragma unroll
    for (int r = 0; r < 8; ++r) {
      float mnew = fmaxf(mrun[r], mblk[r]);
      alpha[r] = __expf(mrun[r] - mnew);
      mrun[r]  = mnew;
      lblk[r]  = 0.f;
    }
#pragma unroll
    for (int t = 0; t < 4; ++t)
#pragma unroll
      for (int r = 0; r < 8; ++r) {
        float p = __expf(s[t][r] - mrun[r]);
        s[t][r] = p;
        lblk[r] += p;
      }
#pragma unroll
    for (int d = 1; d < 16; d <<= 1)
#pragma unroll
      for (int r = 0; r < 8; ++r)
        lblk[r] += __shfl_xor(lblk[r], d, 16);
#pragma unroll
    for (int r = 0; r < 8; ++r) lrun[r] = lrun[r] * alpha[r] + lblk[r];

#pragma unroll
    for (int t = 0; t < 4; ++t)
#pragma unroll
      for (int r = 0; r < 8; ++r)
        pw[(8 * hl + r) * KP + 16 * t + ln] = (_Float16)s[t][r];
    asm volatile("s_wait_dscnt 0" ::: "memory");
    v16h pa0 = ld_afrag(pw + ln * KP, 0, hl);
    v16h pa1 = ld_afrag(pw + ln * KP, 32, hl);

#pragma unroll
    for (int t = 0; t < 4; ++t) {
      const _Float16* vr = &vtsm[(16 * t + ln) * KP];
      v8f c = o[t];
#pragma unroll
      for (int r = 0; r < 8; ++r) c[r] *= alpha[r];
      c = wmma_f16(pa0, ld_bfrag(vr + 16 * hl), c);
      c = wmma_f16(pa1, ld_bfrag(vr + 32 + 16 * hl), c);
      o[t] = c;
    }
  }

#pragma unroll
  for (int r = 0; r < 8; ++r) {
    const size_t row = (size_t)h * SEQ + (size_t)(qb * BLK + w * 16 + 8 * hl + r);
    float inv = 1.0f / lrun[r];
#pragma unroll
    for (int t = 0; t < 4; ++t)
      out[row * DHEAD + 16 * t + ln] = o[t][r] * inv;
  }
}

extern "C" void kernel_launch(void* const* d_in, const int* in_sizes, int n_in,
                              void* d_out, int out_size, void* d_ws, size_t ws_size,
                              hipStream_t stream) {
  const float* q   = (const float*)d_in[0];
  const float* k   = (const float*)d_in[1];
  const float* v   = (const float*)d_in[2];
  const int*   msk = (const int*)d_in[3];
  float* out = (float*)d_out;

  const size_t nelem  = (size_t)HEADS * SEQ * DHEAD;          // 3.15M per tensor
  const size_t f16bytes = nelem * sizeof(_Float16);           // 6.29MB
  dim3 grid(NB, HEADS);                                       // 64 Q blocks x 12 heads

  if (ws_size >= 3 * f16bytes) {
    _Float16* qf = (_Float16*)d_ws;
    _Float16* kf = qf + nelem;
    _Float16* vt = kf + nelem;
    bigbird_prep_f16_kernel<<<(int)((nelem + 255) / 256), 256, 0, stream>>>(q, k, v,
                                                                            qf, kf, vt);
    bigbird_attn_tdm_kernel<<<grid, 128, 0, stream>>>(qf, kf, vt, msk, out);
  } else {
    bigbird_attn_fp32_kernel<<<grid, 128, 0, stream>>>(q, k, v, msk, out);
  }
}
